// EdgeModel_48541720379566
// MI455X (gfx1250) — matrix operand order repaired and checked by reading, not compiled, not run on hardware
//
#include <hip/hip_runtime.h>
#include <cstdint>

typedef __attribute__((ext_vector_type(16))) _Float16 v16h;
typedef __attribute__((ext_vector_type(8)))  _Float16 v8h;
typedef __attribute__((ext_vector_type(2)))  _Float16 v2h;
typedef __attribute__((ext_vector_type(8)))  float    v8f;

#define V_IN   128
#define HIDN   256
#define KT1    14            // 448/32 k-chunks, layer 1
#define KT2    8             // 256/32 k-chunks, layer 2
#define KS_TOT (KT1 + KT2)   // unified 22-step K schedule (W2 follows W1 in ws)
#define NTN    16            // 256/16 n-tiles
#define HSTR   264           // padded f16 row stride for H (conflict-free, 16B-aligned)
#define WAVES  8
#define BLK_T  (WAVES * 32)
#define BTILE  16384         // bytes per staged B k-tile (16 nt * 32 lanes * 32 B)
#define LDS_B  (2 * BTILE)                  // 32768: double buffer
#define LDS_H  (WAVES * 16 * HSTR * 2)      // 67584: per-wave transpose buffers
#define LDS_TOTAL (LDS_B + LDS_H)           // 100352 B

// ---------------------------------------------------------------------------
// Repack [K x 256] f32 weights into f16 WMMA B-fragment order; each k-tile's
// 16 KB is split lo(v0..3)/hi(v4..7) so LDS reads are 16 B/lane stride
// (bank-conflict-free ds_load_b128).
// dword index inside k-tile: half*2048 + (nt*32+lane)*4 + (v&3)
// ---------------------------------------------------------------------------
__global__ void pack_w(const float* __restrict__ W, uint32_t* __restrict__ o, int nkt) {
    int i = blockIdx.x * blockDim.x + threadIdx.x;
    if (i >= nkt * 4096) return;
    int kt   = i >> 12;
    int r    = i & 4095;
    int half = r >> 11;
    int q    = r & 2047;
    int v    = (q & 3) + half * 4;
    int lane = (q >> 2) & 31;
    int nt   = q >> 7;
    int n = nt * 16 + (lane & 15);
    int k = kt * 32 + (lane >> 4) * 8 + half * 8 + 2 * v;
    union { v2h h; uint32_t u; } p;
    p.h = (v2h){ (_Float16)W[k * HIDN + n], (_Float16)W[(k + 1) * HIDN + n] };
    o[i] = p.u;
}

__device__ __forceinline__ v16h cat16(v8h lo, v8h hi) {
    return __builtin_shufflevector(lo, hi, 0,1,2,3,4,5,6,7,8,9,10,11,12,13,14,15);
}

// async DMA one 16 KB B k-tile (ks-th tile of the unified W1|W2 stream) to LDS
__device__ __forceinline__ void stage_tile(uint64_t wbase, int ks, int buf,
                                           uint32_t lds0, int t) {
    #pragma unroll
    for (int j = 0; j < 4; ++j) {
        uint32_t goff = (uint32_t)ks * (uint32_t)BTILE + (uint32_t)t * 16u + (uint32_t)j * 4096u;
        uint32_t loff = lds0 + (uint32_t)buf * (uint32_t)BTILE + (uint32_t)t * 16u + (uint32_t)j * 4096u;
        asm volatile("global_load_async_to_lds_b128 %0, %1, %2"
                     :: "v"(loff), "v"(goff), "s"(wbase) : "memory");
    }
}

__device__ __forceinline__ v16h load_b(const char* bl, int buf, int nt, int lane) {
    const v8h* lo = (const v8h*)(bl + buf * BTILE +        (nt * 32 + lane) * 16);
    const v8h* hi = (const v8h*)(bl + buf * BTILE + 8192 + (nt * 32 + lane) * 16);
    return cat16(*lo, *hi);
}

struct ARaw { float4 q0, q1, q2, q3; };

// issue layer-1 A-fragment loads for k-step kt (consumed one step later)
__device__ __forceinline__ ARaw load_a(int kt, const float* __restrict__ src,
                                       const float* __restrict__ dst,
                                       const float* __restrict__ eat,
                                       const float* __restrict__ u,
                                       long e, long ub, int g) {
    const int col = kt * 32;
    const float* base;
    if      (col < 128) base = src + e * 128 + col;
    else if (col < 256) base = dst + e * 128 + (col - 128);
    else if (col < 320) base = eat + e * 64  + (col - 256);
    else                base = u   + ub * 64 + (col - 320);
    const float4* p = (const float4*)(base + 8 * g);
    ARaw r;
    r.q0 = p[0]; r.q1 = p[1]; r.q2 = p[4]; r.q3 = p[5];   // K: 8g..8g+7, 16+8g..23+8g
    return r;
}

__device__ __forceinline__ v16h cvt_a(const ARaw& r) {
    v16h a;
    a[0]=(_Float16)r.q0.x;  a[1]=(_Float16)r.q0.y;  a[2]=(_Float16)r.q0.z;  a[3]=(_Float16)r.q0.w;
    a[4]=(_Float16)r.q1.x;  a[5]=(_Float16)r.q1.y;  a[6]=(_Float16)r.q1.z;  a[7]=(_Float16)r.q1.w;
    a[8]=(_Float16)r.q2.x;  a[9]=(_Float16)r.q2.y;  a[10]=(_Float16)r.q2.z; a[11]=(_Float16)r.q2.w;
    a[12]=(_Float16)r.q3.x; a[13]=(_Float16)r.q3.y; a[14]=(_Float16)r.q3.z; a[15]=(_Float16)r.q3.w;
    return a;
}

// ---------------------------------------------------------------------------
// 8 waves/WG; wave w owns M-tile blockIdx*8+w. B tiles double-buffered in LDS
// via async DMA (8x less L2 fabric traffic); A prefetched one k-step ahead;
// split cluster/workgroup barrier with private work in the signal->wait gap.
// ---------------------------------------------------------------------------
__global__ __launch_bounds__(BLK_T)
void edge_mlp(const float* __restrict__ src, const float* __restrict__ dst,
              const float* __restrict__ eat, const float* __restrict__ u,
              const int*  __restrict__ batch,
              const uint32_t* __restrict__ wpk,     // packed W1 (14 tiles) | W2 (8 tiles)
              const float* __restrict__ b1, const float* __restrict__ b2,
              float* __restrict__ out, int Etot)
{
    extern __shared__ char lds[];
    char*      Bl = lds;
    const int  t     = threadIdx.x;
    const int  wave  = t >> 5;
    const int  lane  = t & 31;
    const int  g     = lane >> 4;
    const int  idx16 = lane & 15;
    _Float16*  H = (_Float16*)(lds + LDS_B) + wave * 16 * HSTR;

    const long e0 = ((long)blockIdx.x * WAVES + wave) * 16;
    long e = e0 + idx16;
    if (e >= Etot) e = Etot - 1;                  // clamp; stores predicated later
    const long ub = (long)batch[e];

    const uint32_t lds0  = (uint32_t)(uintptr_t)lds;   // flat->LDS: low 32 bits
    const uint64_t wbase = (uint64_t)(uintptr_t)wpk;

    stage_tile(wbase, 0, 0, lds0, t);             // prologue: k-tile 0 -> buf 0
    ARaw araw = load_a(0, src, dst, eat, u, e, ub, g);   // prefetch A for ks=0

    v8f acc[NTN];
    #pragma unroll
    for (int nt = 0; nt < NTN; ++nt) {            // layer-1 bias as accumulator init
        float bv = b1[nt * 16 + idx16];
        #pragma unroll
        for (int r = 0; r < 8; ++r) acc[nt][r] = bv;
    }

    #pragma unroll
    for (int ks = 0; ks < KS_TOT; ++ks) {
        asm volatile("s_wait_asynccnt 0x0" ::: "memory");   // own ks-tile landed
        asm volatile("s_wait_dscnt 0x0"    ::: "memory");   // my ks-1 B reads done
        asm volatile("s_barrier_signal -1" ::: "memory");

        // ---- private work hidden in the barrier gap ----
        v16h a;
        if (ks < KT1) {
            a = cvt_a(araw);                                 // loads issued last step
            if (ks + 1 < KT1)
                araw = load_a(ks + 1, src, dst, eat, u, e, ub, g);
        } else {
            const int kt = ks - KT1;
            const _Float16* ap = &H[idx16 * HSTR + kt * 32 + 8 * g];
            a = cat16(*(const v8h*)ap, *(const v8h*)(ap + 16));  // wave-private LDS
        }

        asm volatile("s_barrier_wait -1"   ::: "memory");   // all ks tiles visible
        if (ks + 1 < KS_TOT)
            stage_tile(wbase, ks + 1, (ks + 1) & 1, lds0, t); // overlap with compute
        const int buf = ks & 1;

        // ---- 16 WMMAs with ping-pong B fragments ----
        v16h b = load_b(Bl, buf, 0, lane);
        #pragma unroll
        for (int nt = 0; nt < NTN; ++nt) {
            v16h bn = b;
            if (nt + 1 < NTN) bn = load_b(Bl, buf, nt + 1, lane);
            acc[nt] = __builtin_amdgcn_wmma_f32_16x16x32_f16(false, a, false, b,
                                                             (short)0, acc[nt], false, false);
            b = bn;
        }

        if (ks == KT1 - 1) {                      // layer boundary: ReLU -> H (private)
            #pragma unroll
            for (int nt = 0; nt < NTN; ++nt) {
                #pragma unroll
                for (int r = 0; r < 8; ++r) {
                    float hv = acc[nt][r];
                    hv = hv > 0.0f ? hv : 0.0f;
                    H[(r + 8 * g) * HSTR + nt * 16 + idx16] = (_Float16)hv;
                }
            }
            #pragma unroll
            for (int nt = 0; nt < NTN; ++nt) {    // re-init with layer-2 bias
                float bv = b2[nt * 16 + idx16];
                #pragma unroll
                for (int r = 0; r < 8; ++r) acc[nt][r] = bv;
            }
        }
    }

    #pragma unroll
    for (int nt = 0; nt < NTN; ++nt) {
        #pragma unroll
        for (int r = 0; r < 8; ++r) {
            long erow = e0 + r + 8 * g;
            if (erow < Etot)
                out[(size_t)erow * HIDN + nt * 16 + idx16] = acc[nt][r];
        }
    }
}

// ---------------------------------------------------------------------------
extern "C" void kernel_launch(void* const* d_in, const int* in_sizes, int n_in,
                              void* d_out, int out_size, void* d_ws, size_t ws_size,
                              hipStream_t stream) {
    const float* src = (const float*)d_in[0];
    const float* dst = (const float*)d_in[1];
    const float* eat = (const float*)d_in[2];
    const float* u   = (const float*)d_in[3];
    const int*   bat = (const int*)d_in[4];
    const float* W1  = (const float*)d_in[5];
    const float* b1  = (const float*)d_in[6];
    const float* W2  = (const float*)d_in[7];
    const float* b2  = (const float*)d_in[8];
    float* out = (float*)d_out;

    const int E = in_sizes[0] / V_IN;             // 500000

    uint32_t* w1p = (uint32_t*)d_ws;              // 14 k-tiles, then W2's 8 k-tiles
    uint32_t* w2p = w1p + KT1 * 4096;

    pack_w<<<(KT1 * 4096 + 255) / 256, 256, 0, stream>>>(W1, w1p, KT1);
    pack_w<<<(KT2 * 4096 + 255) / 256, 256, 0, stream>>>(W2, w2p, KT2);

    (void)hipFuncSetAttribute((const void*)edge_mlp,
                              hipFuncAttributeMaxDynamicSharedMemorySize, LDS_TOTAL);

    const int ntiles = (E + 15) / 16;             // 31250 M-tiles
    const int nblk   = (ntiles + WAVES - 1) / WAVES;   // 3907 workgroups
    edge_mlp<<<nblk, BLK_T, LDS_TOTAL, stream>>>(src, dst, eat, u, bat,
                                                 w1p, b1, b2, out, E);
}
